// VQVAE_45896020525586
// MI455X (gfx1250) — compile-verified, matrix operand order checked by reading, not run on hardware
//
#include <hip/hip_runtime.h>

// ---------------------------------------------------------------------------
// VQ-VAE forward for MI455X (gfx1250).
// All matmuls -> v_wmma_f32_16x16x32_bf16 (wave32, f32 accumulate).
// Generic GEMM: 128x128 (or 128x64) workgroup tile, 8 waves, wave tile 32x64
// (2 A-frags x 4 B-frags = 8 WMMA per 32-deep k-step), double-buffered LDS
// with software pipelining (global->reg fetch of tile k+1 overlapped with the
// WMMA chain of tile k), global_prefetch two tiles ahead.
// Every launch in this model is an exact multiple of the tile sizes
// (M in {128,256,512,8192}, N in {64,512,1024}, K % 32 == 0), so the hot loop
// and epilogue carry no bounds checks.
// ---------------------------------------------------------------------------

typedef __bf16 bhalf;
typedef __attribute__((ext_vector_type(16))) __bf16 v16bh;
typedef __attribute__((ext_vector_type(8)))  __bf16 v8bh;
typedef __attribute__((ext_vector_type(8)))  float  v8f;

#define WG   256
#define BB   16
#define NBIN 256
#define TT   512
#define CL   512
#define NE   1024
#define NH   8
#define HD   64

#define BM  128
#define BK  32
#define LDP 40   // padded LDS row stride (elements) to avoid bank conflicts

// ---------------- fragment load from LDS (row-major tile, stride LDP) -------
// 16-bit A-matrix layout (ISA 7.12.2): lane<16 -> M=lane, K = [0..7],[16..23];
// lane>=16 -> M=lane-16, K = [8..15],[24..31]. B tiles are stored transposed
// in LDS ([N][K]) so the same loader produces the B fragment.
__device__ inline v16bh ld_frag(const bhalf* base) {
  const int lane = threadIdx.x & 31;
  const int r  = lane & 15;
  const int kb = (lane >> 4) << 3;
  const bhalf* p = base + r * LDP + kb;
  v8bh lo = *(const v8bh*)(p);
  v8bh hi = *(const v8bh*)(p + 16);
  v16bh f;
#pragma unroll
  for (int i = 0; i < 8; ++i) { f[i] = lo[i]; f[i + 8] = hi[i]; }
  return f;
}

__device__ inline v8f wmma_bf16(v16bh a, v16bh b, v8f c) {
  return __builtin_amdgcn_wmma_f32_16x16x32_bf16(
      /*neg_a=*/false, a, /*neg_b=*/false, b,
      /*c_mod=*/(short)0, c, /*reuse_a=*/false, /*reuse_b=*/false);
}

// C/D layout: VGPR e -> m = e + (lane>=16 ? 8 : 0); n = lane & 15.
__device__ inline void epi_tile(v8f acc, int mbase, int nbase,
                                int ldc, long long coff,
                                float* Cf, bhalf* Cb,
                                const float* bias, int biasMode, int relu) {
  const int lane  = threadIdx.x & 31;
  const int nloc  = lane & 15;
  const int mhalf = (lane >> 4) << 3;
#pragma unroll
  for (int e = 0; e < 8; ++e) {
    int m = mbase + mhalf + e;
    int n = nbase + nloc;
    float v = acc[e];
    if (biasMode == 1)      v += bias[m];
    else if (biasMode == 2) v += bias[n];
    if (relu) v = fmaxf(v, 0.0f);
    long long ci = coff + (long long)m * ldc + n;
    if (Cf) Cf[ci] = v;
    if (Cb) Cb[ci] = (bhalf)v;
  }
}

// ---------------- generic bf16 WMMA GEMM -----------------------------------
// C[M,N] = A[M,K] @ B[K,N] (+bias)(+relu), batched over grid.z with
// z -> (zb = z/H, zh = z%H) offsets so all B*H attention GEMMs are one launch.
// TRANSB means B is given as [N,K] row-major (QK^T scores, codebook E^T).
template <int BNT, bool TRANSB>
__global__ __launch_bounds__(WG) void k_gemm(
    const bhalf* __restrict__ A, const bhalf* __restrict__ B,
    float* __restrict__ Cf, bhalf* __restrict__ Cb,
    const float* __restrict__ bias,
    int K, int lda, int ldb, int ldc,
    long long sAz, long long sAh, long long sBz, long long sBh,
    long long sCz, long long sCh,
    int H, int biasMode, int relu)
{
  constexpr int NFR = BNT / 32;   // B fragments per wave (2 or 4)
  constexpr int BCH = BNT / 64;   // B staging chunks per thread (1 or 2)

  __shared__ __align__(16) bhalf As[2][BM][LDP];
  __shared__ __align__(16) bhalf Bs[2][BNT][LDP];

  const int tid = threadIdx.x;
  const int z = blockIdx.z;
  const int zb = z / H, zh = z % H;
  A += zb * sAz + zh * sAh;
  B += zb * sBz + zh * sBh;
  const long long coff = zb * sCz + zh * sCh;

  const int n0 = blockIdx.x * BNT;
  const int m0 = blockIdx.y * BM;

  const int w  = tid >> 5;
  const int wm = (w >> 1) << 5;           // 0,32,64,96
  const int wn = (w & 1) * (BNT / 2);     // 0 or BNT/2

  // per-thread staging coordinates (exact-fit: no guards anywhere)
  int arow[2], acol[2];
  const bhalf* aptr[2];
#pragma unroll
  for (int i = 0; i < 2; ++i) {
    int g = tid + i * WG;                 // 512 chunks of 8 bf16 (128x32 tile)
    arow[i] = g >> 2;
    acol[i] = (g & 3) << 3;
    aptr[i] = A + (long long)(m0 + arow[i]) * lda + acol[i];
  }
  int brow[BCH], bcol[BCH];
  const bhalf* bptr[BCH];
#pragma unroll
  for (int i = 0; i < BCH; ++i) {
    int g = tid + i * WG;
    if (TRANSB) {                         // B[N,K]: direct 16B rows
      brow[i] = g >> 2;                   // n
      bcol[i] = (g & 3) << 3;             // k
      bptr[i] = B + (long long)(n0 + brow[i]) * ldb + bcol[i];
    } else {                              // B[K,N]: read 8 n's, scatter to LDS
      brow[i] = g / (BNT / 8);            // k
      bcol[i] = (g % (BNT / 8)) << 3;     // n
      bptr[i] = B + (long long)brow[i] * ldb + n0 + bcol[i];
    }
  }

  const int nk = K / BK;
  v8f acc[2][NFR];
#pragma unroll
  for (int i = 0; i < 2; ++i)
#pragma unroll
    for (int j = 0; j < NFR; ++j) acc[i][j] = v8f{};

  v8bh ra[2], rb[BCH];

  auto fetch = [&](int kt) {
    const long long ko = (long long)kt * BK;
#pragma unroll
    for (int i = 0; i < 2; ++i) ra[i] = *(const v8bh*)(aptr[i] + ko);
#pragma unroll
    for (int i = 0; i < BCH; ++i) {
      const bhalf* pb = TRANSB ? (bptr[i] + ko) : (bptr[i] + ko * ldb);
      rb[i] = *(const v8bh*)(pb);
    }
  };
  auto stage = [&](int buf) {
#pragma unroll
    for (int i = 0; i < 2; ++i)
      *(v8bh*)(&As[buf][arow[i]][acol[i]]) = ra[i];
    if (TRANSB) {
#pragma unroll
      for (int i = 0; i < BCH; ++i)
        *(v8bh*)(&Bs[buf][brow[i]][bcol[i]]) = rb[i];
    } else {
#pragma unroll
      for (int i = 0; i < BCH; ++i)
#pragma unroll
        for (int jj = 0; jj < 8; ++jj)
          Bs[buf][bcol[i] + jj][brow[i]] = rb[i][jj];
    }
  };

  fetch(0);
  stage(0);
  __syncthreads();

  for (int kt = 0; kt < nk; ++kt) {
    const int cur = kt & 1;
    if (kt + 1 < nk) {
      fetch(kt + 1);                       // overlap with WMMA chain below
      if (kt + 2 < nk)
        __builtin_prefetch(aptr[0] + (long long)(kt + 2) * BK, 0, 1);
    }
    v16bh af[2], bfr[NFR];
    af[0] = ld_frag(&As[cur][wm][0]);
    af[1] = ld_frag(&As[cur][wm + 16][0]);
#pragma unroll
    for (int j = 0; j < NFR; ++j) bfr[j] = ld_frag(&Bs[cur][wn + j * 16][0]);
#pragma unroll
    for (int i = 0; i < 2; ++i)
#pragma unroll
      for (int j = 0; j < NFR; ++j)
        acc[i][j] = wmma_bf16(af[i], bfr[j], acc[i][j]);
    if (kt + 1 < nk) stage(cur ^ 1);
    __syncthreads();
  }

#pragma unroll
  for (int i = 0; i < 2; ++i)
#pragma unroll
    for (int j = 0; j < NFR; ++j)
      epi_tile(acc[i][j], m0 + wm + i * 16, n0 + wn + j * 16,
               ldc, coff, Cf, Cb, bias, biasMode, relu);
}

// ---------------- elementwise / reduction kernels --------------------------

__global__ __launch_bounds__(WG) void k_cvt(const float* __restrict__ s,
                                            bhalf* __restrict__ d, int n) {
  int i = blockIdx.x * WG + threadIdx.x, st = gridDim.x * WG;
  for (; i < n; i += st) d[i] = (bhalf)s[i];
}

__global__ __launch_bounds__(WG) void k_scale(const float* __restrict__ x,
                                              const float* __restrict__ wp,
                                              bhalf* __restrict__ o, int n) {
  int i = blockIdx.x * WG + threadIdx.x, st = gridDim.x * WG;
  for (; i < n; i += st) {
    int c = (i / TT) % NBIN;
    o[i] = (bhalf)(x[i] * wp[c]);
  }
}

// col[b][ci*3+k][t] = src[b][ci][t+k-1] (zero padded) -- matches flat conv W
__global__ __launch_bounds__(WG) void k_im2col(const bhalf* __restrict__ src,
                                               bhalf* __restrict__ col,
                                               int Cin, long long total) {
  long long i = (long long)blockIdx.x * WG + threadIdx.x;
  long long st = (long long)gridDim.x * WG;
  for (; i < total; i += st) {
    int t = (int)(i % TT);
    long long r = i / TT;
    int kk = (int)(r % 3); r /= 3;
    int ci = (int)(r % Cin);
    int b  = (int)(r / Cin);
    int ts = t + kk - 1;
    bhalf v = (bhalf)0.0f;
    if (ts >= 0 && ts < TT) v = src[((long long)b * Cin + ci) * TT + ts];
    col[i] = v;
  }
}

// in [B,C,T] f32 -> out [B,T,C] (f32 and/or bf16)
__global__ __launch_bounds__(WG) void k_ct2tc(const float* __restrict__ in,
                                              float* __restrict__ of,
                                              bhalf* __restrict__ ob,
                                              int C, long long total) {
  long long i = (long long)blockIdx.x * WG + threadIdx.x;
  long long st = (long long)gridDim.x * WG;
  for (; i < total; i += st) {
    int t = (int)(i % TT);
    int c = (int)((i / TT) % C);
    int b = (int)(i / ((long long)TT * C));
    float v = in[i];
    long long o = ((long long)b * TT + t) * C + c;
    if (of) of[o] = v;
    if (ob) ob[o] = (bhalf)v;
  }
}

// in [B,T,C] f32 -> out [B,C,T] (f32 and/or bf16)
__global__ __launch_bounds__(WG) void k_tc2ct(const float* __restrict__ in,
                                              float* __restrict__ of,
                                              bhalf* __restrict__ ob,
                                              int C, long long total) {
  long long i = (long long)blockIdx.x * WG + threadIdx.x;
  long long st = (long long)gridDim.x * WG;
  for (; i < total; i += st) {
    int c = (int)(i % C);
    int t = (int)((i / C) % TT);
    int b = (int)(i / ((long long)TT * C));
    float v = in[i];
    long long o = ((long long)b * C + c) * TT + t;
    if (of) of[o] = v;
    if (ob) ob[o] = (bhalf)v;
  }
}

// wave-per-row softmax over len (<=512), bf16 in-place, f32 math
__global__ __launch_bounds__(WG) void k_softmax(bhalf* __restrict__ S,
                                                int rows, int len, float scale) {
  int wv = (blockIdx.x * WG + threadIdx.x) >> 5;
  int lane = threadIdx.x & 31;
  if (wv >= rows) return;
  bhalf* row = S + (long long)wv * len;
  float vals[16];
  int ni = len >> 5;
  float mx = -1e30f;
  for (int i = 0; i < ni; ++i) {
    float v = scale * (float)row[lane + (i << 5)];
    vals[i] = v;
    mx = fmaxf(mx, v);
  }
  for (int o = 16; o > 0; o >>= 1) mx = fmaxf(mx, __shfl_xor(mx, o, 32));
  float sum = 0.0f;
  for (int i = 0; i < ni; ++i) { float e = __expf(vals[i] - mx); vals[i] = e; sum += e; }
  for (int o = 16; o > 0; o >>= 1) sum += __shfl_xor(sum, o, 32);
  float inv = 1.0f / sum;
  for (int i = 0; i < ni; ++i) row[lane + (i << 5)] = (bhalf)(vals[i] * inv);
}

// wave-per-row residual + LayerNorm: ho = LN(h + a)*g + b (also bf16 mirror)
__global__ __launch_bounds__(WG) void k_res_ln(const float* __restrict__ h,
                                               const float* __restrict__ a,
                                               const float* __restrict__ g,
                                               const float* __restrict__ bb,
                                               float* __restrict__ ho,
                                               bhalf* __restrict__ hob,
                                               int rows, int C) {
  int wv = (blockIdx.x * WG + threadIdx.x) >> 5;
  int lane = threadIdx.x & 31;
  if (wv >= rows) return;
  const float* hr = h + (long long)wv * C;
  const float* ar = a + (long long)wv * C;
  float vals[16];
  int ni = C >> 5;
  float s = 0.0f, ss = 0.0f;
  for (int i = 0; i < ni; ++i) {
    int c = lane + (i << 5);
    float v = hr[c] + ar[c];
    vals[i] = v; s += v; ss += v * v;
  }
  for (int o = 16; o > 0; o >>= 1) { s += __shfl_xor(s, o, 32); ss += __shfl_xor(ss, o, 32); }
  float mu = s / (float)C;
  float var = ss / (float)C - mu * mu;
  float rstd = rsqrtf(var + 1e-5f);
  for (int i = 0; i < ni; ++i) {
    int c = lane + (i << 5);
    float o = (vals[i] - mu) * rstd * g[c] + bb[c];
    ho[(long long)wv * C + c] = o;
    hob[(long long)wv * C + c] = (bhalf)o;
  }
}

__global__ __launch_bounds__(WG) void k_enorm(const float* __restrict__ e,
                                              float* __restrict__ en,
                                              int rows, int C) {
  int wv = (blockIdx.x * WG + threadIdx.x) >> 5;
  int lane = threadIdx.x & 31;
  if (wv >= rows) return;
  float s = 0.0f;
  for (int c = lane; c < C; c += 32) {
    float v = e[(long long)wv * C + c];
    s += v * v;
  }
  for (int o = 16; o > 0; o >>= 1) s += __shfl_xor(s, o, 32);
  if (lane == 0) en[wv] = s;
}

// argmin_n (|e_n|^2 - 2*G[r,n]); |z|^2 is constant per row -> same argmin
__global__ __launch_bounds__(WG) void k_argmin(const float* __restrict__ G,
                                               const float* __restrict__ en,
                                               int* __restrict__ idx,
                                               int rows, int N) {
  int wv = (blockIdx.x * WG + threadIdx.x) >> 5;
  int lane = threadIdx.x & 31;
  if (wv >= rows) return;
  const float* row = G + (long long)wv * N;
  float best = 3.4e38f; int bi = 0;
  for (int n = lane; n < N; n += 32) {
    float d = en[n] - 2.0f * row[n];
    if (d < best) { best = d; bi = n; }
  }
  for (int o = 16; o > 0; o >>= 1) {
    float ob = __shfl_xor(best, o, 32);
    int oi = __shfl_xor(bi, o, 32);
    if (ob < best || (ob == best && oi < bi)) { best = ob; bi = oi; }
  }
  if (lane == 0) idx[wv] = bi;
}

// zq: write f32 [B,T,C] (decoder hidden), bf16 mirror, and [B,C,T] output
__global__ __launch_bounds__(WG) void k_gather(const int* __restrict__ idx,
                                               const float* __restrict__ emb,
                                               float* __restrict__ h,
                                               bhalf* __restrict__ hb,
                                               float* __restrict__ zq_ct,
                                               long long total) {
  long long i = (long long)blockIdx.x * WG + threadIdx.x;
  long long st = (long long)gridDim.x * WG;
  for (; i < total; i += st) {
    int c = (int)(i % CL);
    long long bt = i / CL;
    int t = (int)(bt % TT);
    int b = (int)(bt / TT);
    float v = emb[(long long)idx[bt] * CL + c];
    h[i] = v;
    hb[i] = (bhalf)v;
    zq_ct[((long long)b * CL + c) * TT + t] = v;
  }
}

__global__ __launch_bounds__(WG) void k_copy_idx(const int* __restrict__ s,
                                                 int* __restrict__ d, int n) {
  int i = blockIdx.x * WG + threadIdx.x;
  if (i < n) d[i] = s[i];
}

// ---------------------------------------------------------------------------
// Host orchestration
// ---------------------------------------------------------------------------

// input index map (setup_inputs dict order, leaves flattened)
enum {
  IN_X = 0, IN_WPROJ = 1,
  IN_EC0W = 2, IN_EC0B = 3, IN_EC1W = 4, IN_EC1B = 5, IN_EC2W = 6, IN_EC2B = 7,
  IN_EA0 = 8, IN_EA1 = 18, IN_CB = 28, IN_DA0 = 29, IN_DA1 = 39,
  IN_DC0W = 49, IN_DC0B = 50, IN_DC1W = 51, IN_DC1B = 52, IN_DC2W = 53, IN_DC2B = 54,
  IN_REFW = 55, IN_REFB = 56
};
// attn param offsets within a block: +0 wq +1 bq +2 wk +3 bk +4 wv +5 bv
//                                    +6 wo +7 bo +8 ln_g +9 ln_b

extern "C" void kernel_launch(void* const* d_in, const int* in_sizes, int n_in,
                              void* d_out, int out_size, void* d_ws, size_t ws_size,
                              hipStream_t stream) {
  (void)n_in; (void)out_size; (void)ws_size;

  // ---- carve workspace (requires ~225MB) ----
  char* p = (char*)d_ws;
  auto carve = [&](size_t bytes) -> void* {
    void* r = (void*)p;
    p += (bytes + 255) & ~(size_t)255;
    return r;
  };

  bhalf* wec[3]; const int encW[3] = {IN_EC0W, IN_EC1W, IN_EC2W};
  bhalf* wdc[3]; const int decW[3] = {IN_DC0W, IN_DC1W, IN_DC2W};
  for (int i = 0; i < 3; ++i) wec[i] = (bhalf*)carve((size_t)in_sizes[encW[i]] * 2);
  for (int i = 0; i < 3; ++i) wdc[i] = (bhalf*)carve((size_t)in_sizes[decW[i]] * 2);
  bhalf* wat[4][4]; const int ab[4] = {IN_EA0, IN_EA1, IN_DA0, IN_DA1};
  for (int a = 0; a < 4; ++a)
    for (int j = 0; j < 4; ++j)
      wat[a][j] = (bhalf*)carve((size_t)in_sizes[ab[a] + j * 2] * 2);
  bhalf* cbbf   = (bhalf*)carve((size_t)in_sizes[IN_CB] * 2);
  bhalf* refwbf = (bhalf*)carve((size_t)in_sizes[IN_REFW] * 2);

  bhalf* zsc  = (bhalf*)carve((size_t)BB * NBIN * TT * 2);
  bhalf* colb = (bhalf*)carve((size_t)BB * (3 * CL) * TT * 2);   // max Cin=512
  bhalf* cbuf = (bhalf*)carve((size_t)BB * CL * TT * 2);
  float* tmpf = (float*)carve((size_t)BB * TT * CL * 4);
  float* hbuf = (float*)carve((size_t)BB * TT * CL * 4);
  bhalf* hbf  = (bhalf*)carve((size_t)BB * TT * CL * 2);
  bhalf* qb   = (bhalf*)carve((size_t)BB * TT * CL * 2);
  bhalf* kbuf = (bhalf*)carve((size_t)BB * TT * CL * 2);
  bhalf* vbuf = (bhalf*)carve((size_t)BB * TT * CL * 2);
  bhalf* obuf = (bhalf*)carve((size_t)BB * TT * CL * 2);
  bhalf* sbuf = (bhalf*)carve((size_t)BB * NH * TT * TT * 2);    // 67 MB probs
  float* Gbuf = (float*)carve((size_t)BB * TT * NE * 4);         // 33 MB dist
  float* enb  = (float*)carve((size_t)NE * 4);
  int*   idxb = (int*)carve((size_t)BB * TT * 4);

  // ---- output offsets (recon, z, z_q, indices concatenated) ----
  float* out_recon = (float*)d_out;
  float* out_z     = out_recon + (size_t)BB * NBIN * TT;
  float* out_zq    = out_z + (size_t)BB * CL * TT;
  int*   out_idx   = (int*)(out_zq + (size_t)BB * CL * TT);

  const int MR = BB * TT;  // 8192 token rows

  auto cvt = [&](int ii, bhalf* dst) {
    int n = in_sizes[ii];
    int blocks = (n + WG - 1) / WG; if (blocks > 4096) blocks = 4096;
    k_cvt<<<dim3(blocks), dim3(WG), 0, stream>>>((const float*)d_in[ii], dst, n);
  };

  // ---- weight conversion (f32 -> bf16) ----
  for (int i = 0; i < 3; ++i) { cvt(encW[i], wec[i]); cvt(decW[i], wdc[i]); }
  for (int a = 0; a < 4; ++a)
    for (int j = 0; j < 4; ++j) cvt(ab[a] + j * 2, wat[a][j]);
  cvt(IN_CB, cbbf);
  cvt(IN_REFW, refwbf);

  // ---- conv helper: im2col + per-batch GEMM (M=Cout, N=T, K=3*Cin) ----
  auto conv = [&](const bhalf* src, int Cin, int Cout, const bhalf* w,
                  const float* bias, float* outf, bhalf* outb) {
    int K3 = Cin * 3;
    long long total = (long long)BB * K3 * TT;
    int blocks = (int)((total + WG - 1) / WG);
    k_im2col<<<dim3(blocks), dim3(WG), 0, stream>>>(src, colb, Cin, total);
    dim3 g(TT / 128, Cout / 128, BB);
    k_gemm<128, false><<<g, dim3(WG), 0, stream>>>(w, colb, outf, outb, bias,
        K3, K3, TT, TT,
        0, 0, (long long)K3 * TT, 0, (long long)Cout * TT, 0,
        1, 1, 1);
  };

  // ---- attention block (updates hbuf/hbf in place) ----
  auto attn_block = [&](int ib, int a) {
    const float* bqp = (const float*)d_in[ib + 1];
    const float* bkp = (const float*)d_in[ib + 3];
    const float* bvp = (const float*)d_in[ib + 5];
    const float* bop = (const float*)d_in[ib + 7];
    const float* lg  = (const float*)d_in[ib + 8];
    const float* lb  = (const float*)d_in[ib + 9];
    dim3 gp(CL / 128, MR / BM, 1);
    // Q, K, V projections (M=8192, N=K=512), bias per-column
    k_gemm<128, false><<<gp, dim3(WG), 0, stream>>>(hbf, wat[a][0], nullptr, qb, bqp,
        CL, CL, CL, CL, 0, 0, 0, 0, 0, 0, 1, 2, 0);
    k_gemm<128, false><<<gp, dim3(WG), 0, stream>>>(hbf, wat[a][1], nullptr, kbuf, bkp,
        CL, CL, CL, CL, 0, 0, 0, 0, 0, 0, 1, 2, 0);
    k_gemm<128, false><<<gp, dim3(WG), 0, stream>>>(hbf, wat[a][2], nullptr, vbuf, bvp,
        CL, CL, CL, CL, 0, 0, 0, 0, 0, 0, 1, 2, 0);
    // scores = Q @ K^T over all (b,h) in one launch: z = b*NH + h
    dim3 gs(TT / 128, TT / BM, BB * NH);
    k_gemm<128, true><<<gs, dim3(WG), 0, stream>>>(qb, kbuf, nullptr, sbuf, nullptr,
        HD, CL, CL, TT,
        (long long)TT * CL, HD, (long long)TT * CL, HD,
        (long long)TT * TT * NH, (long long)TT * TT,
        NH, 0, 0);
    // softmax rows (scale = 1/sqrt(64))
    k_softmax<<<dim3(BB * NH * TT / 8), dim3(WG), 0, stream>>>(
        sbuf, BB * NH * TT, TT, 0.125f);
    // O = P @ V (N=64 exact-fit instantiation)
    dim3 ga(HD / 64, TT / BM, BB * NH);
    k_gemm<64, false><<<ga, dim3(WG), 0, stream>>>(sbuf, vbuf, nullptr, obuf, nullptr,
        TT, TT, CL, CL,
        (long long)TT * TT * NH, (long long)TT * TT,
        (long long)TT * CL, HD,
        (long long)TT * CL, HD,
        NH, 0, 0);
    // out projection (f32 for residual)
    k_gemm<128, false><<<gp, dim3(WG), 0, stream>>>(obuf, wat[a][3], tmpf, nullptr, bop,
        CL, CL, CL, CL, 0, 0, 0, 0, 0, 0, 1, 2, 0);
    // residual + LayerNorm (in place)
    k_res_ln<<<dim3(MR / 8), dim3(WG), 0, stream>>>(
        hbuf, tmpf, lg, lb, hbuf, hbf, MR, CL);
  };

  // =============================== pipeline ===============================
  // Encoder: per-bin weighting
  {
    int n = BB * NBIN * TT;
    k_scale<<<dim3((n + WG - 1) / WG), dim3(WG), 0, stream>>>(
        (const float*)d_in[IN_X], (const float*)d_in[IN_WPROJ], zsc, n);
  }
  // Encoder convs (relu fused in GEMM epilogue)
  conv(zsc,  NBIN, 128, wec[0], (const float*)d_in[IN_EC0B], nullptr, cbuf);
  conv(cbuf, 128,  256, wec[1], (const float*)d_in[IN_EC1B], nullptr, cbuf);
  conv(cbuf, 256,  512, wec[2], (const float*)d_in[IN_EC2B], tmpf, nullptr);
  {
    long long tot = (long long)BB * CL * TT;
    k_ct2tc<<<dim3((int)((tot + WG - 1) / WG)), dim3(WG), 0, stream>>>(
        tmpf, hbuf, hbf, CL, tot);
  }
  // Encoder attention
  attn_block(IN_EA0, 0);
  attn_block(IN_EA1, 1);
  // z output ([B,T,C] -> [B,C,T])
  {
    long long tot = (long long)BB * TT * CL;
    k_tc2ct<<<dim3((int)((tot + WG - 1) / WG)), dim3(WG), 0, stream>>>(
        hbuf, out_z, nullptr, CL, tot);
  }
  // Codebook quantization
  k_enorm<<<dim3(NE / 8), dim3(WG), 0, stream>>>(
      (const float*)d_in[IN_CB], enb, NE, CL);
  {
    dim3 gc(NE / 128, MR / BM, 1);
    k_gemm<128, true><<<gc, dim3(WG), 0, stream>>>(hbf, cbbf, Gbuf, nullptr, nullptr,
        CL, CL, CL, NE, 0, 0, 0, 0, 0, 0, 1, 0, 0);
  }
  k_argmin<<<dim3(MR / 8), dim3(WG), 0, stream>>>(Gbuf, enb, idxb, MR, NE);
  {
    long long tot = (long long)BB * TT * CL;
    k_gather<<<dim3((int)((tot + WG - 1) / WG)), dim3(WG), 0, stream>>>(
        idxb, (const float*)d_in[IN_CB], hbuf, hbf, out_zq, tot);
  }
  k_copy_idx<<<dim3((MR + WG - 1) / WG), dim3(WG), 0, stream>>>(idxb, out_idx, MR);
  // Decoder attention (hidden = z_q)
  attn_block(IN_DA0, 2);
  attn_block(IN_DA1, 3);
  // Decoder convs: [B,T,C] -> bf16 [B,C,T] then conv stack
  {
    long long tot = (long long)BB * TT * CL;
    k_tc2ct<<<dim3((int)((tot + WG - 1) / WG)), dim3(WG), 0, stream>>>(
        hbuf, nullptr, cbuf, CL, tot);
  }
  conv(cbuf, 512, 256, wdc[0], (const float*)d_in[IN_DC0B], nullptr, cbuf);
  conv(cbuf, 256, 128, wdc[1], (const float*)d_in[IN_DC1B], nullptr, cbuf);
  conv(cbuf, 128, 256, wdc[2], (const float*)d_in[IN_DC2B], nullptr, cbuf);
  // Refine: recon[b,n,t] = sum_m ref_w[n,m]*h[b,m,t] + ref_b[n]
  {
    dim3 gr(TT / 128, NBIN / BM, BB);
    k_gemm<128, false><<<gr, dim3(WG), 0, stream>>>(refwbf, cbuf, out_recon, nullptr,
        (const float*)d_in[IN_REFB],
        NBIN, NBIN, TT, TT,
        0, 0, (long long)NBIN * TT, 0, (long long)NBIN * TT, 0,
        1, 1, 0);
  }
}
